// Gpt2Attention_54417235640661
// MI455X (gfx1250) — matrix operand order repaired and checked by reading, not compiled, not run on hardware
//
#include <hip/hip_runtime.h>

typedef __attribute__((ext_vector_type(8)))  _Float16 v8h;
typedef __attribute__((ext_vector_type(16))) _Float16 v16h;
typedef __attribute__((ext_vector_type(8)))  float    v8f;
typedef __attribute__((ext_vector_type(4)))  unsigned int u32x4;
typedef __attribute__((ext_vector_type(8)))  int      i32x8;
typedef __attribute__((ext_vector_type(4)))  int      i32x4;

#define NB  4
#define NS  2048
#define ND  1024
#define NH  16
#define NHD 64

#if defined(__has_builtin)
#if __has_builtin(__builtin_amdgcn_tensor_load_to_lds)
#define HAVE_TDM 1
#else
#define HAVE_TDM 0
#endif
#else
#define HAVE_TDM 0
#endif

__device__ __forceinline__ v8h ld8(const _Float16* p) { return *(const v8h*)p; }

__device__ __forceinline__ v16h combine(v8h lo, v8h up) {
    v16h r;
#pragma unroll
    for (int i = 0; i < 8; ++i) { r[i] = lo[i]; r[i + 8] = up[i]; }
    return r;
}

__device__ __forceinline__ v8f wmma16(v16h a, v16h b, v8f c) {
    return __builtin_amdgcn_wmma_f32_16x16x32_f16(false, a, false, b, (short)0, c,
                                                  false, false);
}

// A-fragment (16x32 f16): lane m = l16; halves K = 8*hi..8*hi+7 and 16+8*hi..23+8*hi
__device__ __forceinline__ v16h afrag(const _Float16* row, int hi) {
    return combine(ld8(row + 8 * hi), ld8(row + 16 + 8 * hi));
}
// B-fragment (32x16 f16): lane n = l16; halves K = 8*hi..8*hi+7 and 16+8*hi..23+8*hi
__device__ __forceinline__ v16h bfrag(const _Float16* col, int hi) {
    return combine(ld8(col + 8 * hi), ld8(col + 16 + 8 * hi));
}

__device__ __forceinline__ void wait_tensor0() {
#if defined(__has_builtin) && __has_builtin(__builtin_amdgcn_s_wait_tensorcnt)
    __builtin_amdgcn_s_wait_tensorcnt(0);
#else
    asm volatile("s_wait_tensorcnt 0" ::: "memory");
#endif
}

#if HAVE_TDM
// TDM: DMA a 2D f16 tile [128 rows x 32 halves] (row stride 1024 elements)
// into LDS at lds_byte_addr, padding each 64B row with 16B -> LDS stride 40 halves.
__device__ __forceinline__ void tdm_load_tile32(const _Float16* gsrc,
                                                unsigned lds_byte_addr) {
    unsigned long long ga = (unsigned long long)(size_t)gsrc;
    u32x4 g0;
    g0[0] = 1u;                                   // count=1, user load descriptor
    g0[1] = lds_byte_addr;                        // LDS destination (bytes)
    g0[2] = (unsigned)(ga & 0xffffffffu);         // global_addr[31:0]
    g0[3] = (unsigned)((ga >> 32) & 0x01ffffffu)  // global_addr[56:32]
            | (2u << 30);                         // type = 2 ("image")
    i32x8 g1;
    g1[0] = (1 << 16)      // data_size = 2 bytes
          | (1 << 20)      // pad_enable
          | (3 << 22)      // pad_interval: 16 DWORDs (64B)
          | (3 << 25);     // pad_amount:    4 DWORDs (16B)
    g1[1] = (int)(32u << 16);                       // tensor_dim0 = 32
    g1[2] = (int)(128u << 16);                      // tensor_dim1 = 128
    g1[3] = (int)(32u << 16);                       // tile_dim0 = 32
    g1[4] = (int)128;                               // tile_dim1 = 128
    g1[5] = (int)ND;                                // tensor_dim0_stride = 1024
    g1[6] = 0;
    g1[7] = 0;
    i32x4 z4 = {0, 0, 0, 0};
    i32x8 z8 = {0, 0, 0, 0, 0, 0, 0, 0};
    __builtin_amdgcn_tensor_load_to_lds(g0, g1, z4, z4, z8, 0);
}
#endif

// ---------------- weight convert / transpose ----------------
__global__ __launch_bounds__(256) void cvt_wattn(const float* __restrict__ w,
                                                 _Float16* __restrict__ wT) {
    int idx = blockIdx.x * 256 + threadIdx.x;   // idx = n*1024 + k, N = 3072
    int n = idx >> 10, k = idx & 1023;
    wT[idx] = (_Float16)w[(size_t)k * 3072 + n];
}
__global__ __launch_bounds__(256) void cvt_wproj(const float* __restrict__ w,
                                                 _Float16* __restrict__ wT) {
    int idx = blockIdx.x * 256 + threadIdx.x;   // idx = m*1024 + hd
    int m = idx >> 10, hd = idx & 1023;
    wT[idx] = (_Float16)w[(size_t)hd * 1024 + m];
}

// ---------------- QKV GEMM: [8192,1024] x [1024,3072] ----------------
__global__ __launch_bounds__(256) void qkv_gemm(const float* __restrict__ x,
                                                const _Float16* __restrict__ wT,
                                                const float* __restrict__ bias,
                                                _Float16* __restrict__ qb,
                                                _Float16* __restrict__ kb,
                                                _Float16* __restrict__ vtb) {
    __shared__ _Float16 As[128][40];
    __shared__ _Float16 Bs[2][128][40];
    const int tid = threadIdx.x;
    const int lane = tid & 31, l16 = lane & 15, hi = lane >> 4;
    const int wave = tid >> 5, wm = wave >> 1, wn = wave & 1;
    const int m0 = blockIdx.y * 128, n0 = blockIdx.x * 128;

    const int arow = tid >> 3;          // 0..31 (+i*32)
    const int ac4 = (tid & 7) * 4;      // 0..28
    const float* xbase = x + (size_t)m0 * ND + ac4;

    v8f acc[2][4];
#pragma unroll
    for (int i = 0; i < 2; ++i)
#pragma unroll
        for (int j = 0; j < 4; ++j) acc[i][j] = (v8f)0.f;

    const int NKT = ND / 32;
#if HAVE_TDM
    if (wave == 0)
        tdm_load_tile32(wT + (size_t)n0 * ND, (unsigned)(size_t)&Bs[0][0][0]);
#endif
    // Register-pipelined A tile: loads for iteration it+1 issued during
    // iteration it; the s_wait_loadcnt before the cvt is then already satisfied.
    float4 areg[4];
#pragma unroll
    for (int i = 0; i < 4; ++i)
        areg[i] = *(const float4*)(xbase + (size_t)(arow + i * 32) * ND);

    for (int it = 0; it < NKT; ++it) {
        const int kt = it * 32;
        const int buf = it & 1;
#pragma unroll
        for (int i = 0; i < 4; ++i) {               // cvt f32->f16, stage to LDS
            As[arow + i * 32][ac4 + 0] = (_Float16)areg[i].x;
            As[arow + i * 32][ac4 + 1] = (_Float16)areg[i].y;
            As[arow + i * 32][ac4 + 2] = (_Float16)areg[i].z;
            As[arow + i * 32][ac4 + 3] = (_Float16)areg[i].w;
        }
        if (it + 1 < NKT) {
#pragma unroll
            for (int i = 0; i < 4; ++i)             // issue next A-tile loads
                areg[i] = *(const float4*)(xbase + (size_t)(arow + i * 32) * ND + kt + 32);
        }
#if HAVE_TDM
        if (wave == 0) wait_tensor0();
        __syncthreads();
        if (wave == 0 && it + 1 < NKT)
            tdm_load_tile32(wT + (size_t)n0 * ND + kt + 32,
                            (unsigned)(size_t)&Bs[buf ^ 1][0][0]);
#else
#pragma unroll
        for (int i = 0; i < 2; ++i) {               // B: 128 n-rows x 32 k
            int idx = tid + i * 256;
            int row = idx >> 2, kg = (idx & 3) * 8;
            *(v8h*)&Bs[buf][row][kg] =
                *(const v8h*)(wT + (size_t)(n0 + row) * ND + kt + kg);
        }
        __syncthreads();
#endif
        v16h a[2], b[4];
#pragma unroll
        for (int sm = 0; sm < 2; ++sm) a[sm] = afrag(&As[wm * 32 + sm * 16 + l16][0], hi);
#pragma unroll
        for (int sn = 0; sn < 4; ++sn) b[sn] = bfrag(&Bs[buf][wn * 64 + sn * 16 + l16][0], hi);
#pragma unroll
        for (int sm = 0; sm < 2; ++sm)
#pragma unroll
            for (int sn = 0; sn < 4; ++sn) acc[sm][sn] = wmma16(a[sm], b[sn], acc[sm][sn]);
        __syncthreads();
    }

    // Epilogue: t/h/dbase are wave-uniform (derived from scalar tile coords),
    // so the q/k/v split is a scalar branch.
#pragma unroll
    for (int sm = 0; sm < 2; ++sm)
#pragma unroll
        for (int sn = 0; sn < 4; ++sn) {
            const int gm0 = m0 + wm * 32 + sm * 16 + 8 * hi;
            const int cbase = n0 + wn * 64 + sn * 16;        // uniform, 16-aligned
            const int t = cbase >> 10;
            const int h = (cbase & 1023) >> 6;
            const int dbase = cbase & 63;
            const int bb = gm0 >> 11;                        // uniform over r
            const int s0 = gm0 & 2047;
            const int d = dbase + l16;
            const float bv = bias[cbase + l16];
            const size_t bh = (size_t)(bb * NH + h);
            if (t == 0) {
                _Float16* base = qb + (bh * NS + s0) * NHD + d;
#pragma unroll
                for (int r = 0; r < 8; ++r)
                    base[(size_t)r * NHD] = (_Float16)((acc[sm][sn][r] + bv) * 0.125f);
            } else if (t == 1) {
                _Float16* base = kb + (bh * NS + s0) * NHD + d;
#pragma unroll
                for (int r = 0; r < 8; ++r)
                    base[(size_t)r * NHD] = (_Float16)(acc[sm][sn][r] + bv);
            } else {
                v8h pk;                                       // rows contiguous in V^T
#pragma unroll
                for (int r = 0; r < 8; ++r) pk[r] = (_Float16)(acc[sm][sn][r] + bv);
                *(v8h*)(vtb + (bh * NHD + d) * NS + s0) = pk;
            }
        }
}

// ---------------- causal flash attention ----------------
__global__ __launch_bounds__(256) void flash_attn(const _Float16* __restrict__ q,
                                                  const _Float16* __restrict__ k,
                                                  const _Float16* __restrict__ vt,
                                                  _Float16* __restrict__ ao) {
    __shared__ _Float16 P[8][16][40];
    const int tid = threadIdx.x;
    const int lane = tid & 31, l16 = lane & 15, hi = lane >> 4;
    const int wave = tid >> 5;
    const int b = blockIdx.z, h = blockIdx.y;
    const int q0 = blockIdx.x * 128 + wave * 16;
    const size_t bh = (size_t)(b * NH + h);
    const _Float16* qp = q + bh * NS * NHD;
    const _Float16* kp = k + bh * NS * NHD;
    const _Float16* vp = vt + bh * NHD * NS;

    v16h aq[2];
    {
        const _Float16* r = qp + (size_t)(q0 + l16) * NHD;
        aq[0] = afrag(r, hi);
        aq[1] = afrag(r + 32, hi);
    }

    v8f accO[4];
#pragma unroll
    for (int i = 0; i < 4; ++i) accO[i] = (v8f)0.f;
    float mi[8], li[8];
#pragma unroll
    for (int r = 0; r < 8; ++r) { mi[r] = -3.0e38f; li[r] = 0.f; }

    const int nkb = (q0 + 16 + 31) >> 5;
    for (int kb = 0; kb < nkb; ++kb) {
        const int kbase = kb * 32;
        if (kb + 1 < nkb) {
            __builtin_prefetch(kp + (size_t)(kbase + 32 + l16) * NHD, 0, 3);
            __builtin_prefetch(vp + (size_t)l16 * NS + kbase + 32, 0, 3);
        }
        // ---- S = Q K^T (two 16-key tiles, chained over d=64) ----
        v8f sc[2];
#pragma unroll
        for (int jt = 0; jt < 2; ++jt) {
            const _Float16* r = kp + (size_t)(kbase + jt * 16 + l16) * NHD;
            v16h b0 = bfrag(r, hi);
            v16h b1 = bfrag(r + 32, hi);
            v8f s = (v8f)0.f;
            s = wmma16(aq[0], b0, s);
            s = wmma16(aq[1], b1, s);
            sc[jt] = s;
        }
        // ---- causal mask + online softmax ----
        float pr[2][8], alpha[8];
#pragma unroll
        for (int r = 0; r < 8; ++r) {
            int mg = q0 + 8 * hi + r;
            sc[0][r] = (kbase + l16 > mg) ? -1.0e9f : sc[0][r];
            sc[1][r] = (kbase + 16 + l16 > mg) ? -1.0e9f : sc[1][r];
            float t = fmaxf(sc[0][r], sc[1][r]);
#pragma unroll
            for (int off = 1; off < 16; off <<= 1) t = fmaxf(t, __shfl_xor(t, off, 32));
            float mn = fmaxf(mi[r], t);
            alpha[r] = __expf(mi[r] - mn);
            mi[r] = mn;
            float p0 = __expf(sc[0][r] - mn);
            float p1 = __expf(sc[1][r] - mn);
            pr[0][r] = p0; pr[1][r] = p1;
            float rs = p0 + p1;
#pragma unroll
            for (int off = 1; off < 16; off <<= 1) rs += __shfl_xor(rs, off, 32);
            li[r] = li[r] * alpha[r] + rs;
        }
#pragma unroll
        for (int dt = 0; dt < 4; ++dt)
#pragma unroll
            for (int r = 0; r < 8; ++r) accO[dt][r] *= alpha[r];
        // ---- P: C-layout f32 -> LDS -> A-layout f16 (per-wave slab) ----
#pragma unroll
        for (int jt = 0; jt < 2; ++jt)
#pragma unroll
            for (int r = 0; r < 8; ++r)
                P[wave][8 * hi + r][jt * 16 + l16] = (_Float16)pr[jt][r];
        asm volatile("s_wait_dscnt 0" ::: "memory");
        v16h pa = afrag(&P[wave][l16][0], hi);
        // ---- O += P * V (V stored transposed -> contiguous B-frags) ----
#pragma unroll
        for (int dt = 0; dt < 4; ++dt) {
            const _Float16* r = vp + (size_t)(dt * 16 + l16) * NS + kbase;
            accO[dt] = wmma16(pa, bfrag(r, hi), accO[dt]);
        }
    }
    // ---- normalize + store f16 attn_out [B,S,H*64] ----
#pragma unroll
    for (int dt = 0; dt < 4; ++dt) {
        int col = h * NHD + dt * 16 + l16;
#pragma unroll
        for (int r = 0; r < 8; ++r) {
            int mg = q0 + 8 * hi + r;
            ao[((size_t)(b * NS + mg)) * ND + col] = (_Float16)(accO[dt][r] / li[r]);
        }
    }
}

// ---------------- output projection: [8192,1024] x [1024,1024] + bias ----------------
__global__ __launch_bounds__(256) void proj_gemm(const _Float16* __restrict__ a,
                                                 const _Float16* __restrict__ wT,
                                                 const float* __restrict__ bias,
                                                 float* __restrict__ out) {
    __shared__ _Float16 As[2][128][40];
    __shared__ _Float16 Bs[2][128][40];
    const int tid = threadIdx.x;
    const int lane = tid & 31, l16 = lane & 15, hi = lane >> 4;
    const int wave = tid >> 5, wm = wave >> 1, wn = wave & 1;
    const int m0 = blockIdx.y * 128, n0 = blockIdx.x * 128;

    v8f acc[2][4];
#pragma unroll
    for (int i = 0; i < 2; ++i)
#pragma unroll
        for (int j = 0; j < 4; ++j) acc[i][j] = (v8f)0.f;

    const int NKT = ND / 32;
#if HAVE_TDM
    if (wave == 0) {
        tdm_load_tile32(a + (size_t)m0 * ND, (unsigned)(size_t)&As[0][0][0]);
        tdm_load_tile32(wT + (size_t)n0 * ND, (unsigned)(size_t)&Bs[0][0][0]);
    }
#endif
    for (int it = 0; it < NKT; ++it) {
        const int kt = it * 32;
        const int buf = it & 1;
#if HAVE_TDM
        if (wave == 0) wait_tensor0();
        __syncthreads();
        if (wave == 0 && it + 1 < NKT) {
            tdm_load_tile32(a + (size_t)m0 * ND + kt + 32,
                            (unsigned)(size_t)&As[buf ^ 1][0][0]);
            tdm_load_tile32(wT + (size_t)n0 * ND + kt + 32,
                            (unsigned)(size_t)&Bs[buf ^ 1][0][0]);
        }
#else
#pragma unroll
        for (int i = 0; i < 2; ++i) {
            int idx = tid + i * 256;
            int row = idx >> 2, kg = (idx & 3) * 8;
            *(v8h*)&As[buf][row][kg] = *(const v8h*)(a + (size_t)(m0 + row) * ND + kt + kg);
            *(v8h*)&Bs[buf][row][kg] = *(const v8h*)(wT + (size_t)(n0 + row) * ND + kt + kg);
        }
        __syncthreads();
#endif
        v16h af[2], bf[4];
#pragma unroll
        for (int sm = 0; sm < 2; ++sm) af[sm] = afrag(&As[buf][wm * 32 + sm * 16 + l16][0], hi);
#pragma unroll
        for (int sn = 0; sn < 4; ++sn) bf[sn] = bfrag(&Bs[buf][wn * 64 + sn * 16 + l16][0], hi);
#pragma unroll
        for (int sm = 0; sm < 2; ++sm)
#pragma unroll
            for (int sn = 0; sn < 4; ++sn) acc[sm][sn] = wmma16(af[sm], bf[sn], acc[sm][sn]);
        __syncthreads();
    }

#pragma unroll
    for (int sm = 0; sm < 2; ++sm)
#pragma unroll
        for (int sn = 0; sn < 4; ++sn) {
            int gm0 = m0 + wm * 32 + sm * 16 + 8 * hi;
            int c = n0 + wn * 64 + sn * 16 + l16;
            float bv = bias[c];
#pragma unroll
            for (int r = 0; r < 8; ++r)
                out[(size_t)(gm0 + r) * ND + c] = acc[sm][sn][r] + bv;
        }
}

extern "C" void kernel_launch(void* const* d_in, const int* in_sizes, int n_in,
                              void* d_out, int out_size, void* d_ws, size_t ws_size,
                              hipStream_t stream) {
    const float* x      = (const float*)d_in[0];
    const float* w_attn = (const float*)d_in[1];
    const float* b_attn = (const float*)d_in[2];
    const float* w_proj = (const float*)d_in[3];
    const float* b_proj = (const float*)d_in[4];

    char* ws = (char*)d_ws;
    _Float16* wqkvT  = (_Float16*)(ws);                 // 3072*1024*2  = 6,291,456
    _Float16* wprojT = (_Float16*)(ws + 6291456);       // 1024*1024*2  = 2,097,152
    _Float16* qb     = (_Float16*)(ws + 8388608);       // 16 MiB
    _Float16* kb     = (_Float16*)(ws + 25165824);      // 16 MiB
    _Float16* vtb    = (_Float16*)(ws + 41943040);      // 16 MiB
    _Float16* ao     = (_Float16*)(ws + 58720256);      // 16 MiB

    cvt_wattn<<<12288, 256, 0, stream>>>(w_attn, wqkvT);
    cvt_wproj<<<4096, 256, 0, stream>>>(w_proj, wprojT);
    qkv_gemm<<<dim3(24, 64), 256, 0, stream>>>(x, wqkvT, b_attn, qb, kb, vtb);
    flash_attn<<<dim3(16, 16, 4), 256, 0, stream>>>(qb, kb, vtb, ao);
    proj_gemm<<<dim3(8, 64), 256, 0, stream>>>(ao, wprojT, b_proj, (float*)d_out);
}
